// SelfAttention_79370995630539
// MI455X (gfx1250) — matrix operand order repaired and checked by reading, not compiled, not run on hardware
//
#include <hip/hip_runtime.h>
#include <hip/hip_bf16.h>
#include <math.h>

typedef __bf16 bf16_t;
typedef __attribute__((ext_vector_type(16))) __bf16 v16bf;
typedef __attribute__((ext_vector_type(8)))  __bf16 v8bf;
typedef __attribute__((ext_vector_type(8)))  float  v8f;
typedef uint32_t u32x4 __attribute__((ext_vector_type(4)));
typedef uint32_t u32x8 __attribute__((ext_vector_type(8)));

constexpr int kB  = 16;
constexpr int kC  = 512;
constexpr int kCI = 64;
constexpr int kCH = 256;
constexpr int kN  = 64 * 64;   // 4096
constexpr int kP  = kN / 4;    // 1024

__device__ __forceinline__ v8f zero_v8f() {
  v8f z;
#pragma unroll
  for (int i = 0; i < 8; ++i) z[i] = 0.0f;
  return z;
}

__device__ __forceinline__ v16bf cat8(v8bf lo, v8bf hi) {
  v16bf r;
#pragma unroll
  for (int i = 0; i < 8; ++i) { r[i] = lo[i]; r[i + 8] = hi[i]; }
  return r;
}

__device__ __forceinline__ v8f wmma_bf16(v16bf a, v16bf b, v8f c) {
  // D = A(16x32 bf16) * B(32x16 bf16) + C(16x16 f32)
  return __builtin_amdgcn_wmma_f32_16x16x32_bf16(false, a, false, b, (short)0, c,
                                                 false, false);
}

// LDS byte offset of a __shared__ object (generic addr low 32 bits == LDS offset)
__device__ __forceinline__ uint32_t ldsoff(const void* p) {
  return (uint32_t)(uintptr_t)p;
}

// TDM: async 2-D tile load Global -> LDS (bf16 elements, groups 2/3 omitted).
// Rows of tile_d0 elements are packed contiguously into LDS (pitch = tile_d0*2B).
__device__ __forceinline__ void tdm_load_2d(uint32_t lds_byte_off, const void* gaddr,
                                            uint32_t tensor_d0, uint32_t tensor_d1,
                                            uint32_t tile_d0, uint32_t tile_d1,
                                            uint32_t stride0_elems) {
  const uint64_t ga = (uint64_t)(uintptr_t)gaddr;
  u32x4 g0;
  g0[0] = 1u;                                   // count=1, user descriptor
  g0[1] = lds_byte_off;                         // D#.lds_addr
  g0[2] = (uint32_t)ga;                         // global_addr[31:0]
  g0[3] = (uint32_t)(ga >> 32) | 0x80000000u;   // global_addr[56:32] | type=2
  u32x8 g1;
  g1[0] = (1u << 16);                           // data_size = 2 bytes
  g1[1] = (tensor_d0 & 0xFFFFu) << 16;          // tensor_dim0[15:0]
  g1[2] = ((tensor_d0 >> 16) & 0xFFFFu) | ((tensor_d1 & 0xFFFFu) << 16);
  g1[3] = ((tensor_d1 >> 16) & 0xFFFFu) | (tile_d0 << 16);   // tile_dim0
  g1[4] = tile_d1;                              // tile_dim1 (tile_dim2 = 0)
  g1[5] = stride0_elems;                        // tensor_dim0_stride[31:0]
  g1[6] = 0;
  g1[7] = 0;
  asm volatile("tensor_load_to_lds %0, %1" :: "s"(g0), "s"(g1) : "memory");
}

// ---------------------------------------------------------------------------
// Kernel 1: x [B,C,N] f32  ->  xT [B,N,C] bf16 (LDS-tiled transpose)
// ---------------------------------------------------------------------------
__global__ __launch_bounds__(256) void prep_x(const float* __restrict__ x,
                                              bf16_t* __restrict__ xT) {
  __shared__ bf16_t t[32][33];
  const int b = blockIdx.z;
  const int n0 = blockIdx.x * 32;
  const int c0 = blockIdx.y * 32;
  const float* xb = x + (size_t)b * kC * kN;
#pragma unroll
  for (int cc = threadIdx.y; cc < 32; cc += 8)
    t[cc][threadIdx.x] = (bf16_t)xb[(size_t)(c0 + cc) * kN + n0 + threadIdx.x];
  __syncthreads();
  bf16_t* xo = xT + (size_t)b * kN * kC;
#pragma unroll
  for (int nn = threadIdx.y; nn < 32; nn += 8)
    xo[(size_t)(n0 + nn) * kC + c0 + threadIdx.x] = t[threadIdx.x][nn];
}

// ---------------------------------------------------------------------------
// Kernel 2: weights -> bf16.  Wcat = [Wf;Wg;Wh] (384x512), Wvb = Wv (512x256)
// ---------------------------------------------------------------------------
__global__ __launch_bounds__(256) void wprep(const float* __restrict__ Wf,
                                             const float* __restrict__ Wg,
                                             const float* __restrict__ Wh,
                                             const float* __restrict__ Wv,
                                             bf16_t* __restrict__ Wcat,
                                             bf16_t* __restrict__ Wvb) {
  const int i = blockIdx.x * 256 + threadIdx.x;
  if (i < 384 * kC) {
    const int m = i >> 9, k = i & 511;
    float v;
    if (m < 64)        v = Wf[m * kC + k];
    else if (m < 128)  v = Wg[(m - 64) * kC + k];
    else               v = Wh[(m - 128) * kC + k];
    Wcat[i] = (bf16_t)v;
  }
  const int j = i - 384 * kC;
  if (j >= 0 && j < kC * kCH) Wvb[j] = (bf16_t)Wv[j];
}

// ---------------------------------------------------------------------------
// Kernel 3: Y = Wcat @ X  (384 x 4096 per batch).
//  rows   0- 63 -> f   : fbuf[b][n][ci]        (packed v8bf stores)
//  rows  64-127 -> g   : 2x2 maxpool -> gpool[b][p][ci]
//  rows 128-383 -> h   : 2x2 maxpool -> hT[b][ch][p]
// ---------------------------------------------------------------------------
__global__ __launch_bounds__(256) void conv_fgh(const bf16_t* __restrict__ xT,
                                                const bf16_t* __restrict__ Wcat,
                                                bf16_t* __restrict__ fbuf,
                                                bf16_t* __restrict__ gpool,
                                                bf16_t* __restrict__ hT) {
  __shared__ bf16_t ytile[64][256];
  const int b   = blockIdx.z;
  const int mb  = blockIdx.y;            // 0..5
  const int nb  = blockIdx.x;            // 0..15
  const int w   = threadIdx.x >> 5;
  const int ln  = threadIdx.x & 31;
  const int l16 = ln & 15;
  const int hf  = ln >> 4;
  const int wm  = w >> 1, wn = w & 1;
  const int m0  = mb * 64 + wm * 16;
  const int n0  = nb * 256 + wn * 128;
  const bf16_t* xb = xT + (size_t)b * kN * kC;

  v8f acc[8];
#pragma unroll
  for (int t = 0; t < 8; ++t) acc[t] = zero_v8f();

  const bf16_t* arow = Wcat + (size_t)(m0 + l16) * kC;
#pragma unroll 2
  for (int kk = 0; kk < kC; kk += 32) {
    v16bf a = *(const v16bf*)(arow + kk + 16 * hf);
    v16bf bm[8];
#pragma unroll
    for (int t = 0; t < 8; ++t)
      bm[t] = *(const v16bf*)(xb + (size_t)(n0 + t * 16 + l16) * kC + kk + 16 * hf);
    if (kk + 32 < kC)
      __builtin_prefetch((const void*)(xb + (size_t)(n0 + l16) * kC + kk + 32), 0, 1);
#pragma unroll
    for (int t = 0; t < 8; ++t)
      acc[t] = wmma_bf16(a, bm[t], acc[t]);
  }

  if (mb == 0) {  // f region: C-layout rows r=0..7 are stride-1 in [n][ci]
#pragma unroll
    for (int t = 0; t < 8; ++t) {
      v8bf pk;
#pragma unroll
      for (int r = 0; r < 8; ++r) pk[r] = (bf16_t)acc[t][r];
      const int n = n0 + t * 16 + l16;
      *(v8bf*)(fbuf + ((size_t)b * kN + n) * kCI + wm * 16 + 8 * hf) = pk;
    }
  } else {        // g/h region: stage to LDS, fused 2x2 maxpool
#pragma unroll
    for (int t = 0; t < 8; ++t)
#pragma unroll
      for (int r = 0; r < 8; ++r)
        ytile[wm * 16 + r + 8 * hf][wn * 128 + t * 16 + l16] = (bf16_t)acc[t][r];
    __syncthreads();
    for (int i = threadIdx.x; i < 4096; i += 256) {
      const int px = i & 31, py = (i >> 5) & 1, cl = i >> 6;
      const int nloc = py * 128 + (px << 1);
      float mx = fmaxf(fmaxf((float)ytile[cl][nloc], (float)ytile[cl][nloc + 1]),
                       fmaxf((float)ytile[cl][nloc + 64], (float)ytile[cl][nloc + 65]));
      const int p    = (nb * 2 + py) * 32 + px;
      const int grow = mb * 64 + cl;
      if (grow < 128)
        gpool[((size_t)b * kP + p) * kCI + (grow - 64)] = (bf16_t)mx;
      else
        hT[((size_t)b * kCH + (grow - 128)) * kP + p] = (bf16_t)mx;
    }
  }
}

// ---------------------------------------------------------------------------
// Kernel 4: flash attention with TDM-staged key/value chunks.
//  Per chunk (64 keys): wave0 issues tensor_load_to_lds for the NEXT chunk
//  (hv tile [256ch x 64k] + g tile [64k x 64ci], double buffered), waits
//  TENSORcnt, then all 8 waves compute:
//   S = f @ g^T (8 WMMA), online softmax (shfl_xor), probs->LDS->A-layout,
//   O += P @ hv (32 WMMA) into 16 f32 tiles.
// ---------------------------------------------------------------------------
__global__ __launch_bounds__(256) void attn_kernel(const bf16_t* __restrict__ fbuf,
                                                   const bf16_t* __restrict__ gpool,
                                                   const bf16_t* __restrict__ hT,
                                                   bf16_t* __restrict__ vbuf) {
  __shared__ bf16_t hvt[2][kCH][64];    // [ch][key]   64 KB, TDM-filled
  __shared__ bf16_t gld[2][64][kCI];    // [key][ci]   16 KB, TDM-filled
  __shared__ bf16_t lds_p[8][16][64];   // per-wave probs scratch  16 KB
  const int b   = blockIdx.y;
  const int w   = threadIdx.x >> 5;
  const int ln  = threadIdx.x & 31;
  const int l16 = ln & 15, hf = ln >> 4;
  const int q0  = blockIdx.x * 128 + w * 16;

  const bf16_t* gb = gpool + (size_t)b * kP * kCI;
  const bf16_t* hb = hT + (size_t)b * kCH * kP;

  // A-fragments of f: rows q0..q0+15, K = ci 0..63
  const bf16_t* fr = fbuf + ((size_t)b * kN + q0 + l16) * kCI;
  const v16bf af0 = *(const v16bf*)(fr + 16 * hf);
  const v16bf af1 = *(const v16bf*)(fr + 32 + 16 * hf);

  // prologue: stage chunk 0
  if (w == 0) {
    tdm_load_2d(ldsoff(&hvt[0][0][0]), hb, kP, kCH, 64, kCH, kP);
    tdm_load_2d(ldsoff(&gld[0][0][0]), gb, kCI, kP, kCI, 64, kCI);
  }

  v8f o[16];
#pragma unroll
  for (int ct = 0; ct < 16; ++ct) o[ct] = zero_v8f();
  float m_r[8], l_r[8];
#pragma unroll
  for (int r = 0; r < 8; ++r) { m_r[r] = -INFINITY; l_r[r] = 0.0f; }

  for (int ic = 0; ic < 16; ++ic) {
    const int pc  = ic * 64;
    const int cur = ic & 1;
    if (w == 0) {
      if (ic < 15) {   // prefetch next chunk into the other buffer
        const int nx = cur ^ 1;
        tdm_load_2d(ldsoff(&hvt[nx][0][0]), hb + pc + 64, kP, kCH, 64, kCH, kP);
        tdm_load_2d(ldsoff(&gld[nx][0][0]), gb + (size_t)(pc + 64) * kCI,
                    kCI, kP, kCI, 64, kCI);
        __builtin_amdgcn_s_wait_tensorcnt(2);   // in-order: chunk ic complete
      } else {
        __builtin_amdgcn_s_wait_tensorcnt(0);
      }
    }
    __syncthreads();   // chunk ic visible to all waves

    // ---- logits: S[16 x 64] from LDS g tile, K = CI = 64 -> 8 WMMAs ----
    v8f s[4];
#pragma unroll
    for (int t = 0; t < 4; ++t) {
      s[t] = zero_v8f();
      const bf16_t* g0 = &gld[cur][t * 16 + l16][16 * hf];
      s[t] = wmma_bf16(af0, *(const v16bf*)g0, s[t]);
      s[t] = wmma_bf16(af1, *(const v16bf*)(g0 + 32), s[t]);
    }
    // ---- online softmax; rows live across lanes of a half-wave ----
#pragma unroll
    for (int r = 0; r < 8; ++r) {
      float cm = fmaxf(fmaxf(s[0][r], s[1][r]), fmaxf(s[2][r], s[3][r]));
      cm = fmaxf(cm, __shfl_xor(cm, 1, 32));
      cm = fmaxf(cm, __shfl_xor(cm, 2, 32));
      cm = fmaxf(cm, __shfl_xor(cm, 4, 32));
      cm = fmaxf(cm, __shfl_xor(cm, 8, 32));
      const float mnew = fmaxf(m_r[r], cm);
      const float corr = __expf(m_r[r] - mnew);
      float rs = 0.0f;
#pragma unroll
      for (int t = 0; t < 4; ++t) {
        const float pv = __expf(s[t][r] - mnew);
        rs += pv;
        lds_p[w][r + 8 * hf][t * 16 + l16] = (bf16_t)pv;
      }
      rs += __shfl_xor(rs, 1, 32);
      rs += __shfl_xor(rs, 2, 32);
      rs += __shfl_xor(rs, 4, 32);
      rs += __shfl_xor(rs, 8, 32);
      l_r[r] = l_r[r] * corr + rs;
      m_r[r] = mnew;
#pragma unroll
      for (int ct = 0; ct < 16; ++ct) o[ct][r] *= corr;
    }
    asm volatile("s_wait_dscnt 0" ::: "memory");   // per-wave LDS store->load
    // ---- probs C-layout -> A-layout via LDS ----
    const bf16_t* pb = &lds_p[w][l16][0];
    const v16bf ap0 = *(const v16bf*)(pb + 16 * hf);
    const v16bf ap1 = *(const v16bf*)(pb + 32 + 16 * hf);
    // ---- O += P @ hv from LDS hv tile : 16 ch-tiles x 2 WMMAs ----
#pragma unroll
    for (int ct = 0; ct < 16; ++ct) {
      const bf16_t* h0 = &hvt[cur][ct * 16 + l16][16 * hf];
      o[ct] = wmma_bf16(ap0, *(const v16bf*)h0, o[ct]);
      o[ct] = wmma_bf16(ap1, *(const v16bf*)(h0 + 32), o[ct]);
    }
    __syncthreads();   // all reads of buffer `cur` done before TDM reuses it
  }
  // ---- normalize and write v [b][n][ch] bf16 ----
#pragma unroll
  for (int r = 0; r < 8; ++r) {
    const float inv = 1.0f / l_r[r];
    const int q = q0 + r + 8 * hf;
    bf16_t* vr = vbuf + ((size_t)b * kN + q) * kCH;
#pragma unroll
    for (int ct = 0; ct < 16; ++ct)
      vr[ct * 16 + l16] = (bf16_t)(o[ct][r] * inv);
  }
}

// ---------------------------------------------------------------------------
// Kernel 5: out = x + beta * (Wv @ V).  Same tiling as conv_fgh, f32 epilogue.
// ---------------------------------------------------------------------------
__global__ __launch_bounds__(256) void conv_v(const bf16_t* __restrict__ vbuf,
                                              const bf16_t* __restrict__ Wvb,
                                              const float* __restrict__ x,
                                              const float* __restrict__ beta,
                                              float* __restrict__ out) {
  const int b   = blockIdx.z;
  const int mbk = blockIdx.y;            // 0..7
  const int nb  = blockIdx.x;            // 0..15
  const int w   = threadIdx.x >> 5;
  const int ln  = threadIdx.x & 31;
  const int l16 = ln & 15, hf = ln >> 4;
  const int wm  = w >> 1, wn = w & 1;
  const int m0  = mbk * 64 + wm * 16;
  const int n0  = nb * 256 + wn * 128;
  const bf16_t* vb = vbuf + (size_t)b * kN * kCH;

  v8f acc[8];
#pragma unroll
  for (int t = 0; t < 8; ++t) acc[t] = zero_v8f();

  const bf16_t* arow = Wvb + (size_t)(m0 + l16) * kCH;
#pragma unroll 2
  for (int kk = 0; kk < kCH; kk += 32) {
    v16bf a = *(const v16bf*)(arow + kk + 16 * hf);
    v16bf bm[8];
#pragma unroll
    for (int t = 0; t < 8; ++t)
      bm[t] = *(const v16bf*)(vb + (size_t)(n0 + t * 16 + l16) * kCH + kk + 16 * hf);
#pragma unroll
    for (int t = 0; t < 8; ++t)
      acc[t] = wmma_bf16(a, bm[t], acc[t]);
  }
  const float bv = beta[0];
#pragma unroll
  for (int t = 0; t < 8; ++t) {
    const int n = n0 + t * 16 + l16;
#pragma unroll
    for (int r = 0; r < 8; ++r) {
      const int m = m0 + r + 8 * hf;
      const size_t idx = ((size_t)b * kC + m) * (size_t)kN + n;
      out[idx] = x[idx] + bv * acc[t][r];
    }
  }
}

// ---------------------------------------------------------------------------
extern "C" void kernel_launch(void* const* d_in, const int* in_sizes, int n_in,
                              void* d_out, int out_size, void* d_ws, size_t ws_size,
                              hipStream_t stream) {
  const float* x    = (const float*)d_in[0];
  const float* Wf   = (const float*)d_in[1];
  const float* Wg   = (const float*)d_in[2];
  const float* Wh   = (const float*)d_in[3];
  const float* Wv   = (const float*)d_in[4];
  const float* beta = (const float*)d_in[5];

  char* ws = (char*)d_ws;
  bf16_t* xT   = (bf16_t*)ws; ws += (size_t)kB * kN * kC * 2;    // 64 MB
  bf16_t* Wcat = (bf16_t*)ws; ws += (size_t)384 * kC * 2;
  bf16_t* Wvb  = (bf16_t*)ws; ws += (size_t)kC * kCH * 2;
  bf16_t* fbuf = (bf16_t*)ws; ws += (size_t)kB * kN * kCI * 2;   // 8 MB
  bf16_t* gpl  = (bf16_t*)ws; ws += (size_t)kB * kP * kCI * 2;   // 2 MB
  bf16_t* hTb  = (bf16_t*)ws; ws += (size_t)kB * kCH * kP * 2;   // 8 MB
  bf16_t* vbuf = (bf16_t*)ws; ws += (size_t)kB * kN * kCH * 2;   // 32 MB

  prep_x<<<dim3(kN / 32, kC / 32, kB), dim3(32, 8), 0, stream>>>(x, xT);
  const int wtot = 384 * kC + kC * kCH;
  wprep<<<(wtot + 255) / 256, 256, 0, stream>>>(Wf, Wg, Wh, Wv, Wcat, Wvb);
  conv_fgh<<<dim3(kN / 256, 6, kB), 256, 0, stream>>>(xT, Wcat, fbuf, gpl, hTb);
  attn_kernel<<<dim3(kN / 128, kB), 256, 0, stream>>>(fbuf, gpl, hTb, vbuf);
  conv_v<<<dim3(kN / 256, kC / 64, kB), 256, 0, stream>>>(vbuf, Wvb, x, beta,
                                                          (float*)d_out);
}
